// ODE_RNN_76785425318663
// MI455X (gfx1250) — matrix operand order repaired and checked by reading, not compiled, not run on hardware
//
#include <hip/hip_runtime.h>
#include <hip/hip_bf16.h>

// ODE-RNN persistent kernel for MI455X (gfx1250, wave32, WMMA).
//
// Each workgroup owns a 16-row batch tile (WMMA M=16) and runs ALL 64
// timesteps with only workgroup barriers (the recurrence is independent
// per batch row). fp32 hidden state lives in registers in the WMMA C/D
// layout; a bf16 shadow copy in LDS feeds the WMMA A operand. Weights are
// converted to bf16 and pre-transposed once so a lane's two B tiles are 64
// contiguous bytes, streamed from L2 as 4 global_load_b128 per k-tile pair.
//
// GEMM k-loops: rolled (#pragma unroll 1) modulo-2 software pipeline with
// LOOP-CARRIED pointers. Rolled + pointer-carried addressing (including in
// the epilogue) keeps every load induction-variant so MachineLICM cannot
// hoist weight tiles across the Euler loop and spill them to scratch.

#define SEQ     64
#define I_DIM   256
#define H_DIM   1024
#define NODE    4
#define TILE_B  16
#define THREADS 1024   // 32 waves -> 4 SIMD32s x 8 waves per WGP

typedef __attribute__((ext_vector_type(16))) __bf16        v16bf;
typedef __attribute__((ext_vector_type(8)))  float         v8f;
typedef __attribute__((ext_vector_type(4)))  unsigned int  u32x4;

union BfPack { u32x4 q[2]; v16bf v; };

static __device__ __forceinline__ unsigned short f2bf(float f) {
  unsigned u = __float_as_uint(f);
  u += 0x7FFFu + ((u >> 16) & 1u);          // round-to-nearest-even
  return (unsigned short)(u >> 16);
}

// Branchless tanh: tanh(v) = 1 - 2/(exp(2v)+1).
// exp2 overflow -> inf -> rcp -> 0 -> +1; underflow -> 0 -> rcp(1) -> -1.
static __device__ __forceinline__ float fast_tanh(float v) {
  float e = __builtin_amdgcn_exp2f(v * 2.8853900817779268f); // 2*log2(e)
  return 1.0f - 2.0f * __builtin_amdgcn_rcpf(e + 1.0f);
}

// 16 contiguous halves at p (two b128 loads).
static __device__ __forceinline__ v16bf ld16(const unsigned short* p) {
  BfPack pk;
  pk.q[0] = *(const u32x4*)(p);
  pk.q[1] = *(const u32x4*)(p + 8);
  return pk.v;
}

// WMMA A operand chunks for this lane: p[0..7] and p[16..23] halves.
static __device__ __forceinline__ v16bf ld_a(const unsigned short* p) {
  BfPack pk;
  pk.q[0] = *(const u32x4*)(p);
  pk.q[1] = *(const u32x4*)(p + 16);
  return pk.v;
}

static __device__ __forceinline__ v8f wmma_bf16(v16bf a, v16bf b, v8f c) {
  return __builtin_amdgcn_wmma_f32_16x16x32_bf16(false, a, false, b,
                                                 (short)0, c, false, false);
}

// acc{0,1} += A[16xK] * Wt[Kx(nb0..+15, nb0+16..+31)], K multiple of 64.
// A layout (16-bit 16x32): lane L<16 -> row L, K chunks [0..7],[16..23];
// lane L>=16 -> row L-16, chunks [8..15],[24..31]  => per-lane base
// asrc + (L&15)*astride + (L>>4)*8, chunks at +0 and +16.
// B layout (16-bit 32x16): lane L -> Wt row kb+L, 16 halves per tile
// => per-lane base Wt + L*H + nb0; tile0 at +0, tile1 at +16 halves.
static __device__ __forceinline__ void gemm_acc(const unsigned short* asrc,
                                                int astride, int K,
                                                const unsigned short* __restrict__ Wt,
                                                int lane, int nb0,
                                                v8f& acc0, v8f& acc1) {
  const unsigned short* ap = asrc + (lane & 15) * astride + (lane >> 4) * 8;
  const unsigned short* bp = Wt + (size_t)lane * H_DIM + nb0;

  v16bf a0  = ld_a(ap);
  v16bf w00 = ld16(bp);
  v16bf w10 = ld16(bp + 16);
  #pragma unroll 1
  for (int it = 0; it < K / 64 - 1; ++it) {
    v16bf a1  = ld_a(ap + 32);
    v16bf w01 = ld16(bp + 32 * H_DIM);
    v16bf w11 = ld16(bp + 32 * H_DIM + 16);
    acc0 = wmma_bf16(a0, w00, acc0);
    acc1 = wmma_bf16(a0, w10, acc1);
    ap += 64;
    bp += (size_t)64 * H_DIM;
    a0  = ld_a(ap);
    w00 = ld16(bp);
    w10 = ld16(bp + 16);
    acc0 = wmma_bf16(a1, w01, acc0);
    acc1 = wmma_bf16(a1, w11, acc1);
  }
  // Last two k-tiles: addresses derive from the loop-carried ap/bp phi,
  // so these loads cannot be hoisted above (and spilled across) the loop.
  v16bf a1  = ld_a(ap + 32);
  v16bf w01 = ld16(bp + 32 * H_DIM);
  v16bf w11 = ld16(bp + 32 * H_DIM + 16);
  acc0 = wmma_bf16(a0, w00, acc0);
  acc1 = wmma_bf16(a0, w10, acc1);
  acc0 = wmma_bf16(a1, w01, acc0);
  acc1 = wmma_bf16(a1, w11, acc1);
}

// One-time: convert fp32 weights to bf16, packed per 32x16 block so lane L's 16 halves hold column n=L%16, k in WMMA B order.
__global__ void convert_weights(const float* __restrict__ W_ode,
                                const float* __restrict__ W_h,
                                const float* __restrict__ W_in,
                                unsigned short* __restrict__ ws) {
  unsigned short* Wt_ode = ws;
  unsigned short* Wt_h   = ws + (size_t)H_DIM * H_DIM;
  unsigned short* Wt_in  = ws + (size_t)2 * H_DIM * H_DIM;
  const size_t total_hh = (size_t)H_DIM * H_DIM;
  const size_t total_in = (size_t)I_DIM * H_DIM;
  const size_t total    = 2 * total_hh + total_in;
  for (size_t idx = (size_t)blockIdx.x * blockDim.x + threadIdx.x;
       idx < total; idx += (size_t)gridDim.x * blockDim.x) {
    if (idx < total_hh) {
      size_t r = idx / H_DIM, c = idx % H_DIM, L = r & 31, s = c & 15, h = L >> 4;
      size_t k = (r & ~(size_t)31) + (s < 8 ? 8 * h + s : 16 + 8 * h + s - 8);
      size_t n = (c & ~(size_t)15) + (L & 15);
      Wt_ode[idx] = f2bf(W_ode[n * H_DIM + k]);
    } else if (idx < 2 * total_hh) {
      size_t j = idx - total_hh;
      size_t r = j / H_DIM, c = j % H_DIM, L = r & 31, s = c & 15, h = L >> 4;
      size_t k = (r & ~(size_t)31) + (s < 8 ? 8 * h + s : 16 + 8 * h + s - 8);
      size_t n = (c & ~(size_t)15) + (L & 15);
      Wt_h[j] = f2bf(W_h[n * H_DIM + k]);
    } else {
      size_t j = idx - 2 * total_hh;
      size_t r = j / H_DIM, c = j % H_DIM, L = r & 31, s = c & 15, h = L >> 4;
      size_t k = (r & ~(size_t)31) + (s < 8 ? 8 * h + s : 16 + 8 * h + s - 8);
      size_t n = (c & ~(size_t)15) + (L & 15);
      Wt_in[j] = f2bf(W_in[n * I_DIM + k]);
    }
  }
}

__global__ __launch_bounds__(THREADS)
void ode_rnn_persistent(const float* __restrict__ x,
                        const float* __restrict__ t,
                        const float* __restrict__ b_in,
                        const float* __restrict__ b_h,
                        const float* __restrict__ b_ode,
                        const unsigned short* __restrict__ Wt_ode,
                        const unsigned short* __restrict__ Wt_h,
                        const unsigned short* __restrict__ Wt_in,
                        float* __restrict__ out) {
  __shared__ unsigned short hbf[TILE_B * H_DIM];   // 32 KB bf16 hidden state
  __shared__ unsigned short xbf[TILE_B * I_DIM];   //  8 KB bf16 x_s tile

  const int tid  = threadIdx.x;
  const int lane = tid & 31;
  const int wave = tid >> 5;
  const int b0   = blockIdx.x * TILE_B;

  const int nb0   = wave * 32;            // this wave owns columns [nb0, nb0+32)
  const int ncol0 = nb0 + (lane & 15);
  const int ncol1 = nb0 + 16 + (lane & 15);
  const int mrow  = (lane >> 4) * 8;      // C/D layout: VGPR d -> row mrow+d

  const float bo0  = b_ode[ncol0];
  const float bo1  = b_ode[ncol1];
  const float bih0 = b_in[ncol0] + b_h[ncol0];
  const float bih1 = b_in[ncol1] + b_h[ncol1];

  // fp32 hidden state in registers, WMMA C/D layout (16x32 slab per wave).
  v8f hst0 = {};
  v8f hst1 = {};

  for (int i = tid; i < TILE_B * H_DIM; i += THREADS) hbf[i] = 0;
  __syncthreads();

  float t_prev = t[0];
  #pragma unroll 1
  for (int s = 0; s < SEQ; ++s) {
    // ---- stage x_s tile (fp32 global -> bf16 LDS) -----------------------
    #pragma unroll
    for (int i = tid; i < TILE_B * I_DIM; i += THREADS) {
      int r = i >> 8, c = i & (I_DIM - 1);
      xbf[i] = f2bf(x[(((size_t)(b0 + r)) * SEQ + s) * I_DIM + c]);
    }
    const float t_cur = t[s];
    const float dth   = (t_cur - t_prev) * (1.0f / NODE);
    t_prev = t_cur;
    __syncthreads();

    // ---- N_ODE explicit Euler steps: h += dth * tanh(h W_ode^T + b) -----
    #pragma unroll 1
    for (int e = 0; e < NODE; ++e) {
      v8f acc0 = {};
      v8f acc1 = {};
      gemm_acc(hbf, H_DIM, H_DIM, Wt_ode, lane, nb0, acc0, acc1);
      __syncthreads();   // all waves finished reading hbf
      #pragma unroll
      for (int d = 0; d < 8; ++d) {
        hst0[d] += dth * fast_tanh(acc0[d] + bo0);
        hst1[d] += dth * fast_tanh(acc1[d] + bo1);
      }
      #pragma unroll
      for (int d = 0; d < 8; ++d) {
        hbf[(mrow + d) * H_DIM + ncol0] = f2bf(hst0[d]);
        hbf[(mrow + d) * H_DIM + ncol1] = f2bf(hst1[d]);
      }
      __syncthreads();   // new hbf visible before next read
    }

    // ---- RNN update: h = tanh(x W_in^T + h W_h^T + b_in + b_h) ----------
    {
      v8f acc0 = {};
      v8f acc1 = {};
      gemm_acc(hbf, H_DIM, H_DIM, Wt_h,  lane, nb0, acc0, acc1);
      gemm_acc(xbf, I_DIM, I_DIM, Wt_in, lane, nb0, acc0, acc1);
      __syncthreads();
      #pragma unroll
      for (int d = 0; d < 8; ++d) {
        hst0[d] = fast_tanh(acc0[d] + bih0);
        hst1[d] = fast_tanh(acc1[d] + bih1);
      }
      #pragma unroll
      for (int d = 0; d < 8; ++d) {
        hbf[(mrow + d) * H_DIM + ncol0] = f2bf(hst0[d]);
        hbf[(mrow + d) * H_DIM + ncol1] = f2bf(hst1[d]);
      }
      __syncthreads();
    }
  }

  // ---- write final h (fp32) out[B, H] -----------------------------------
  #pragma unroll
  for (int d = 0; d < 8; ++d) {
    out[(size_t)(b0 + mrow + d) * H_DIM + ncol0] = hst0[d];
    out[(size_t)(b0 + mrow + d) * H_DIM + ncol1] = hst1[d];
  }
}

extern "C" void kernel_launch(void* const* d_in, const int* in_sizes, int n_in,
                              void* d_out, int out_size, void* d_ws, size_t ws_size,
                              hipStream_t stream) {
  // setup_inputs order: x, t, W_in, b_in, W_h, b_h, W_ode, b_ode (all fp32)
  const float* x     = (const float*)d_in[0];
  const float* t     = (const float*)d_in[1];
  const float* W_in  = (const float*)d_in[2];
  const float* b_in  = (const float*)d_in[3];
  const float* W_h   = (const float*)d_in[4];
  const float* b_h   = (const float*)d_in[5];
  const float* W_ode = (const float*)d_in[6];
  const float* b_ode = (const float*)d_in[7];

  unsigned short* wsp = (unsigned short*)d_ws;  // 4.5 MB of bf16 weights
  unsigned short* Wt_ode = wsp;
  unsigned short* Wt_h   = wsp + (size_t)H_DIM * H_DIM;
  unsigned short* Wt_in  = wsp + (size_t)2 * H_DIM * H_DIM;

  convert_weights<<<2048, 256, 0, stream>>>(W_ode, W_h, W_in, wsp);

  ode_rnn_persistent<<<256 / TILE_B, THREADS, 0, stream>>>(
      x, t, b_in, b_h, b_ode, Wt_ode, Wt_h, Wt_in, (float*)d_out);
}
